// Bogomol_89919435309160
// MI455X (gfx1250) — compile-verified, matrix operand order checked by reading, not run on hardware
//
#include <hip/hip_runtime.h>
#include <hip/hip_bf16.h>

// ---------------- problem dims ----------------
#define BN     16
#define CIN    3
#define HH     224
#define WWD    224
#define HWN    (HH*WWD)          // 50176
#define COUTN  96
#define HIDN   512
#define NHN    4
#define DHN    128
#define SEQN   196               // 14*14
#define NHW    14
#define PDIMN  24576             // 96*16*16
#define CDIMN  864               // 96*9
#define MROWS  (BN*SEQN)         // 3136

typedef __attribute__((ext_vector_type(16))) __bf16 v16bf;
typedef __attribute__((ext_vector_type(8)))  float  v8f;

union FragU { v16bf v; uint4 q[2]; };

// Load a 16x16x32 bf16 WMMA A/B fragment for an NT (K-contiguous) operand.
// lane layout: element e holds K = (e/8)*16 + hi*8 + (e%8) relative to k0.
__device__ __forceinline__ v16bf load_frag(const __hip_bfloat16* rowp, int k0, int hi) {
    FragU f;
    const uint4* p = reinterpret_cast<const uint4*>(rowp + k0 + (hi << 3));
    f.q[0] = p[0];   // K = k0 + hi*8 + [0..7]
    f.q[1] = p[2];   // K = k0 + 16 + hi*8 + [0..7]
    return f.v;
}
__device__ __forceinline__ v16bf zero_frag() {
    FragU f; f.q[0] = make_uint4(0,0,0,0); f.q[1] = make_uint4(0,0,0,0); return f.v;
}
__device__ __forceinline__ v8f wmma_bf16(v16bf a, v16bf b, v8f c) {
    return __builtin_amdgcn_wmma_f32_16x16x32_bf16(false, a, false, b, (short)0, c, false, false);
}

__device__ __forceinline__ float geluf(float x) {
    return 0.5f * x * (1.0f + erff(x * 0.70710678118654752f));
}

__device__ __forceinline__ float breduce_sum(float v, float* red, int n) {
    int t = threadIdx.x;
    red[t] = v; __syncthreads();
    for (int s = n >> 1; s > 0; s >>= 1) { if (t < s) red[t] += red[t + s]; __syncthreads(); }
    float r = red[0]; __syncthreads();
    return r;
}

// ---------------- 1. GroupNorm(1) stats per sample ----------------
__global__ void gn_stats_k(const float* __restrict__ x, float* __restrict__ stats) {
    int b = blockIdx.x, t = threadIdx.x;
    const float* xb = x + (size_t)b * CIN * HWN;
    float s = 0.f, s2 = 0.f;
    for (int i = t; i < CIN * HWN; i += 256) { float v = xb[i]; s += v; s2 += v * v; }
    __shared__ float red[256];
    s  = breduce_sum(s,  red, 256);
    s2 = breduce_sum(s2, red, 256);
    if (t == 0) {
        float inv = 1.f / (float)(CIN * HWN);
        float mu = s * inv, var = s2 * inv - mu * mu;
        stats[b * 2] = mu; stats[b * 2 + 1] = rsqrtf(var + 1e-5f);
    }
}

// ---------------- 2. f32 -> bf16 cast ----------------
__global__ void cast_bf16_k(const float* __restrict__ src, __hip_bfloat16* __restrict__ dst, int n) {
    int i = blockIdx.x * 256 + threadIdx.x;
    if (i < n) dst[i] = __float2bfloat16(src[i]);
}

// ---------------- 3. input conv 3x3 (3->96) + GELU, channels-last bf16 out ----------------
__global__ void conv_in_k(const float* __restrict__ x, const float* __restrict__ stats,
                          const float* __restrict__ gn_g, const float* __restrict__ gn_b,
                          const float* __restrict__ conv_w, const float* __restrict__ conv_b,
                          __hip_bfloat16* __restrict__ featsCL) {
    __shared__ float wsm[COUTN * 27];
    __shared__ float bsm[COUTN];
    int t = threadIdx.x;
    for (int i = t; i < COUTN * 27; i += 256) wsm[i] = conv_w[i];
    if (t < COUTN) bsm[t] = conv_b[t];
    __syncthreads();
    int idx = blockIdx.x * 256 + t;          // over B*HW
    int b = idx / HWN, p = idx % HWN;
    int y = p / WWD, xx = p % WWD;
    float mu = stats[b * 2], rs = stats[b * 2 + 1];
    float v[27];
    #pragma unroll
    for (int ci = 0; ci < 3; ++ci) {
        float g = gn_g[ci] * rs;
        float bb = gn_b[ci] - mu * g;        // xn = x*g + bb ; pad value = 0
        #pragma unroll
        for (int ky = 0; ky < 3; ++ky)
        #pragma unroll
        for (int kx = 0; kx < 3; ++kx) {
            int yy = y + ky - 1, xs = xx + kx - 1;
            float val = 0.f;
            if (yy >= 0 && yy < HH && xs >= 0 && xs < WWD)
                val = x[((size_t)b * CIN + ci) * HWN + yy * WWD + xs] * g + bb;
            v[ci * 9 + ky * 3 + kx] = val;
        }
    }
    __hip_bfloat16* outp = featsCL + (size_t)idx * COUTN;
    for (int o = 0; o < COUTN; ++o) {
        float a = bsm[o];
        #pragma unroll
        for (int k = 0; k < 27; ++k) a += wsm[o * 27 + k] * v[k];
        outp[o] = __float2bfloat16(geluf(a));
    }
}

// ---------------- 4. repack channels-last feats -> flat patches [B,SEQ,24576] bf16 ----------------
__global__ void flatp_k(const __hip_bfloat16* __restrict__ featsCL, __hip_bfloat16* __restrict__ flatp) {
    int idx = blockIdx.x * 256 + threadIdx.x;       // over B*SEQ*PDIM
    int p = idx % PDIMN;
    int bs = idx / PDIMN;
    int s = bs % SEQN, b = bs / SEQN;
    int c = p >> 8, r = p & 255, ph = r >> 4, pw = r & 15;
    int i = s / NHW, j = s % NHW;
    int y = i * 16 + ph, xx = j * 16 + pw;
    flatp[idx] = featsCL[((size_t)b * HWN + y * WWD + xx) * COUTN + c];
}

// ---------------- 5. generic NT WMMA GEMM: C[M,N] = A[M,K] * B[N,K]^T (+bias, act) ----------------
// one wave per block, 32x64 C tile (2 M-frags x 4 N-frags -> 8 WMMAs / k-step).
// M must be a multiple of 32. act: 0 = none, 1 = gelu
__global__ void gemm_nt_k(const __hip_bfloat16* __restrict__ A, const __hip_bfloat16* __restrict__ Bw,
                          float* __restrict__ C, const float* __restrict__ bias,
                          int M, int N, int K, int act) {
    int lane = threadIdx.x, l = lane & 15, hi = lane >> 4;
    int n0 = blockIdx.x * 64, m0 = blockIdx.y * 32;
    const __hip_bfloat16* arow0 = A + (size_t)(m0 + l) * K;
    const __hip_bfloat16* arow1 = A + (size_t)(m0 + 16 + l) * K;
    const __hip_bfloat16* brow[4];
    #pragma unroll
    for (int t = 0; t < 4; ++t) {
        int nr = n0 + t * 16 + l;
        brow[t] = Bw + (size_t)(nr < N ? nr : 0) * K;
    }
    __builtin_prefetch(arow0, 0, 3);
    __builtin_prefetch(arow1, 0, 3);
    #pragma unroll
    for (int t = 0; t < 4; ++t) __builtin_prefetch(brow[t], 0, 3);
    v8f acc[2][4] = {};
    for (int k0 = 0; k0 < K; k0 += 32) {
        v16bf a0 = load_frag(arow0, k0, hi);
        v16bf a1 = load_frag(arow1, k0, hi);
        #pragma unroll
        for (int t = 0; t < 4; ++t) {
            v16bf b = load_frag(brow[t], k0, hi);
            acc[0][t] = wmma_bf16(a0, b, acc[0][t]);
            acc[1][t] = wmma_bf16(a1, b, acc[1][t]);
        }
    }
    #pragma unroll
    for (int mi = 0; mi < 2; ++mi)
    #pragma unroll
    for (int t = 0; t < 4; ++t) {
        int col = n0 + t * 16 + l;
        if (col < N) {
            float bi = bias ? bias[col] : 0.f;
            #pragma unroll
            for (int r = 0; r < 8; ++r) {
                int row = m0 + mi * 16 + r + hi * 8;
                float v = acc[mi][t][r] + bi;
                if (act == 1) v = geluf(v);
                C[(size_t)row * N + col] = v;
            }
        }
    }
}

// ---------------- 6. LayerNorm + tanh positional add; writes f32 h and bf16 h ----------------
__global__ void ln_pos_k(const float* __restrict__ E, const float* __restrict__ ln_g,
                         const float* __restrict__ ln_b, const float* __restrict__ pos_w,
                         const float* __restrict__ pos_b, float* __restrict__ HB,
                         __hip_bfloat16* __restrict__ HBH) {
    int row = blockIdx.x, t = threadIdx.x;     // 256 threads, 2 elems each
    const float* e = E + (size_t)row * HIDN;
    float v0 = e[t], v1 = e[t + 256];
    __shared__ float red[256];
    float m = breduce_sum(v0 + v1, red, 256) * (1.f / HIDN);
    float d0 = v0 - m, d1 = v1 - m;
    float var = breduce_sum(d0 * d0 + d1 * d1, red, 256) * (1.f / HIDN);
    float rs = rsqrtf(var + 1e-5f);
    int s = row % SEQN;
    float ci = (float)(s / NHW) * (1.f / NHW);
    float cj = (float)(s % NHW) * (1.f / NHW);
    #pragma unroll
    for (int e2 = 0; e2 < 2; ++e2) {
        int h = t + e2 * 256;
        float d = e2 ? d1 : d0;
        float pos = tanhf(ci * pos_w[h * 2] + cj * pos_w[h * 2 + 1] + pos_b[h]);
        float hv = d * rs * ln_g[h] + ln_b[h] + pos;
        HB[(size_t)row * HIDN + h] = hv;
        HBH[(size_t)row * HIDN + h] = __float2bfloat16(hv);
    }
}

// ---------------- 7. attention: one block per (query, head, sample), dh=128 ----------------
__global__ void attn_k(const float* __restrict__ Q, const float* __restrict__ K,
                       const float* __restrict__ V, float* __restrict__ O) {
    int qi = blockIdx.x, n = blockIdx.y, b = blockIdx.z, t = threadIdx.x;  // 128 threads
    __shared__ float qs[DHN];
    __shared__ float sc[SEQN];
    __shared__ float red[128];
    size_t base = (size_t)b * SEQN * HIDN + n * DHN;
    qs[t] = Q[base + (size_t)qi * HIDN + t];
    __syncthreads();
    for (int j = t; j < SEQN; j += 128) {
        const float* kr = K + base + (size_t)j * HIDN;
        float acc = 0.f;
        #pragma unroll 8
        for (int d = 0; d < DHN; ++d) acc += qs[d] * kr[d];
        sc[j] = acc * 0.08838834764831845f;   // 1/sqrt(128)
    }
    __syncthreads();
    float mx = -1e30f;
    for (int j = t; j < SEQN; j += 128) mx = fmaxf(mx, sc[j]);
    red[t] = mx; __syncthreads();
    for (int s = 64; s > 0; s >>= 1) { if (t < s) red[t] = fmaxf(red[t], red[t + s]); __syncthreads(); }
    mx = red[0]; __syncthreads();
    float sm = 0.f;
    for (int j = t; j < SEQN; j += 128) { float ex = __expf(sc[j] - mx); sc[j] = ex; sm += ex; }
    sm = breduce_sum(sm, red, 128);
    float inv = 1.f / sm;
    float acc = 0.f;
    for (int j = 0; j < SEQN; ++j) acc += sc[j] * V[base + (size_t)j * HIDN + t];
    O[base + (size_t)qi * HIDN + t] = acc * inv;
}

// ---------------- 8. votes = mean_s tanh(att @ be_w^T + be_b) ----------------
__global__ void votes_k(const float* __restrict__ ATT, const float* __restrict__ be_w,
                        const float* __restrict__ be_b, float* __restrict__ votes) {
    int k9 = blockIdx.x, b = blockIdx.y, t = threadIdx.x;
    float val = 0.f;
    if (t < SEQN) {
        const float* a = ATT + (size_t)(b * SEQN + t) * HIDN;
        const float* w = be_w + k9 * HIDN;
        float acc = be_b[k9];
        for (int h = 0; h < HIDN; ++h) acc += a[h] * w[h];
        val = tanhf(acc);
    }
    __shared__ float red[256];
    float s = breduce_sum(val, red, 256);
    if (t == 0) votes[b * 9 + k9] = s * (1.f / SEQN);
}

// ---------------- 9. repm[b,d] = mean_s R[b,s,d] ----------------
__global__ void repm_k(const float* __restrict__ R, float* __restrict__ repm) {
    int idx = blockIdx.x * 256 + threadIdx.x;
    if (idx >= BN * CDIMN) return;
    int b = idx / CDIMN, d = idx % CDIMN;
    float s = 0.f;
    for (int q = 0; q < SEQN; ++q) s += R[((size_t)(b * SEQN + q)) * CDIMN + d];
    repm[idx] = s * (1.f / SEQN);
}

// ---------------- 10. filter synthesis, repacked to [b][o][(kh*3+kw)*96+ci] bf16 ----------------
__global__ void filt_k(const float* __restrict__ repm, const float* __restrict__ votes,
                       const float* __restrict__ basis, __hip_bfloat16* __restrict__ filtA) {
    int idx = blockIdx.x * 256 + threadIdx.x;
    if (idx >= BN * COUTN * CDIMN) return;
    int w = idx % CDIMN;
    int kidx = w / COUTN, ci = w % COUTN;
    int o = (idx / CDIMN) % COUTN;
    int b = idx / (CDIMN * COUTN);
    float acc = 0.f;
    #pragma unroll
    for (int k9 = 0; k9 < 9; ++k9)
        acc += repm[b * CDIMN + o * 9 + k9] * votes[b * 9 + k9] * basis[k9 * CDIMN + ci * 9 + kidx];
    filtA[idx] = __float2bfloat16(acc);
}

// ---------------- 11. dynamic 96->96 3x3 conv as implicit WMMA GEMM + alpha*feats ----------------
// 32x64 tile per wave: 2 filter-row frags vs 4 spatial frags -> 8 WMMAs / k-step.
__global__ void dynconv_k(const __hip_bfloat16* __restrict__ featsCL,
                          const __hip_bfloat16* __restrict__ filtA,
                          const float* __restrict__ alpha_p, float* __restrict__ out) {
    int lane = threadIdx.x, l = lane & 15, hi = lane >> 4;
    int b = blockIdx.z, m0 = blockIdx.y * 32, n0 = blockIdx.x * 64;
    const __hip_bfloat16* arow0 = filtA + ((size_t)b * COUTN + m0 + l) * CDIMN;
    const __hip_bfloat16* arow1 = filtA + ((size_t)b * COUTN + m0 + 16 + l) * CDIMN;
    __builtin_prefetch(arow0, 0, 3);
    __builtin_prefetch(arow1, 0, 3);
    int px[4], py[4];
    #pragma unroll
    for (int t = 0; t < 4; ++t) { int n = n0 + t * 16 + l; py[t] = n / WWD; px[t] = n % WWD; }
    v8f acc[2][4] = {};
    for (int kidx = 0; kidx < 9; ++kidx) {
        int dy = kidx / 3 - 1, dx = kidx % 3 - 1;
        for (int cc = 0; cc < 3; ++cc) {          // 96 channels = 3 k-steps of 32
            int k0 = kidx * COUTN + cc * 32;
            v16bf a0 = load_frag(arow0, k0, hi);
            v16bf a1 = load_frag(arow1, k0, hi);
            #pragma unroll
            for (int t = 0; t < 4; ++t) {
                int yy = py[t] + dy, xx = px[t] + dx;
                v16bf bf;
                if (yy >= 0 && yy < HH && xx >= 0 && xx < WWD)
                    bf = load_frag(featsCL + ((size_t)b * HWN + yy * WWD + xx) * COUTN, cc * 32, hi);
                else
                    bf = zero_frag();
                acc[0][t] = wmma_bf16(a0, bf, acc[0][t]);
                acc[1][t] = wmma_bf16(a1, bf, acc[1][t]);
            }
        }
    }
    float alpha = alpha_p[0];
    #pragma unroll
    for (int mi = 0; mi < 2; ++mi)
    #pragma unroll
    for (int t = 0; t < 4; ++t) {
        int n = n0 + t * 16 + l;
        #pragma unroll
        for (int r = 0; r < 8; ++r) {
            int o = m0 + mi * 16 + r + hi * 8;
            float f = __bfloat162float(featsCL[((size_t)b * HWN + n) * COUTN + o]);
            out[((size_t)b * COUTN + o) * HWN + n] = alpha * f + acc[mi][t][r];
        }
    }
}

// ---------------- 12. SE: channel means of y ----------------
__global__ void se_mean_k(const float* __restrict__ y, float* __restrict__ smean) {
    int bc = blockIdx.x, t = threadIdx.x;
    const float* p = y + (size_t)bc * HWN;
    float s = 0.f;
    for (int i = t; i < HWN; i += 256) s += p[i];
    __shared__ float red[256];
    s = breduce_sum(s, red, 256);
    if (t == 0) smean[bc] = s * (1.f / HWN);
}

// ---------------- 13. SE gate vector ----------------
__global__ void se_vec_k(const float* __restrict__ smean, const float* __restrict__ w1,
                         const float* __restrict__ b1, const float* __restrict__ w2,
                         const float* __restrict__ b2, float* __restrict__ svec) {
    int b = blockIdx.x, t = threadIdx.x;   // 128 threads
    __shared__ float m[COUTN];
    __shared__ float s1[24];
    if (t < COUTN) m[t] = smean[b * COUTN + t];
    __syncthreads();
    if (t < 24) {
        float a = b1[t];
        for (int c = 0; c < COUTN; ++c) a += m[c] * w1[t * COUTN + c];
        s1[t] = fmaxf(a, 0.f);
    }
    __syncthreads();
    if (t < COUTN) {
        float a = b2[t];
        #pragma unroll
        for (int k = 0; k < 24; ++k) a += s1[k] * w2[t * 24 + k];
        svec[b * COUTN + t] = 1.f / (1.f + __expf(-a));
    }
}

// ---------------- 14. final scale ----------------
__global__ void scale_k(float* __restrict__ y, const float* __restrict__ svec) {
    size_t idx = (size_t)blockIdx.x * 256 + threadIdx.x;
    y[idx] *= svec[idx / HWN];
}

// =====================================================================
extern "C" void kernel_launch(void* const* d_in, const int* in_sizes, int n_in,
                              void* d_out, int out_size, void* d_ws, size_t ws_size,
                              hipStream_t stream) {
    const float* x      = (const float*)d_in[0];
    const float* gn_g   = (const float*)d_in[1];
    const float* gn_b   = (const float*)d_in[2];
    const float* conv_w = (const float*)d_in[3];
    const float* conv_b = (const float*)d_in[4];
    const float* pos_w  = (const float*)d_in[5];
    const float* pos_b  = (const float*)d_in[6];
    const float* comp_w = (const float*)d_in[7];
    const float* comp_b = (const float*)d_in[8];
    const float* ln_g   = (const float*)d_in[9];
    const float* ln_b   = (const float*)d_in[10];
    const float* wq = (const float*)d_in[11]; const float* bq = (const float*)d_in[12];
    const float* wk = (const float*)d_in[13]; const float* bk = (const float*)d_in[14];
    const float* wv = (const float*)d_in[15]; const float* bv = (const float*)d_in[16];
    const float* wo = (const float*)d_in[17]; const float* bo = (const float*)d_in[18];
    const float* be_w = (const float*)d_in[19]; const float* be_b = (const float*)d_in[20];
    const float* patch_basis = (const float*)d_in[21];
    const float* re_w = (const float*)d_in[22]; const float* re_b = (const float*)d_in[23];
    const float* se_w1 = (const float*)d_in[24]; const float* se_b1 = (const float*)d_in[25];
    const float* se_w2 = (const float*)d_in[26]; const float* se_b2 = (const float*)d_in[27];
    const float* alpha = (const float*)d_in[28];
    float* out = (float*)d_out;

    // ---- workspace carve-out (all offsets 256B aligned) ----
    char* w = (char*)d_ws;
    size_t off = 0;
    auto carve = [&](size_t bytes) -> char* {
        char* p = w + off;
        off += (bytes + 255) & ~(size_t)255;
        return p;
    };
    float*           stats   = (float*)carve(BN * 2 * sizeof(float));
    __hip_bfloat16*  featsCL = (__hip_bfloat16*)carve((size_t)BN * HWN * COUTN * 2);
    __hip_bfloat16*  flatp   = (__hip_bfloat16*)carve((size_t)MROWS * PDIMN * 2);
    __hip_bfloat16*  cwh     = (__hip_bfloat16*)carve((size_t)HIDN * PDIMN * 2);
    __hip_bfloat16*  wqh     = (__hip_bfloat16*)carve((size_t)HIDN * HIDN * 2);
    __hip_bfloat16*  wkh     = (__hip_bfloat16*)carve((size_t)HIDN * HIDN * 2);
    __hip_bfloat16*  wvh     = (__hip_bfloat16*)carve((size_t)HIDN * HIDN * 2);
    __hip_bfloat16*  woh     = (__hip_bfloat16*)carve((size_t)HIDN * HIDN * 2);
    __hip_bfloat16*  rewh    = (__hip_bfloat16*)carve((size_t)CDIMN * HIDN * 2);
    float*           E       = (float*)carve((size_t)MROWS * HIDN * 4);
    float*           HB      = (float*)carve((size_t)MROWS * HIDN * 4);
    __hip_bfloat16*  HBH     = (__hip_bfloat16*)carve((size_t)MROWS * HIDN * 2);
    float*           Qb      = (float*)carve((size_t)MROWS * HIDN * 4);
    float*           Kb      = (float*)carve((size_t)MROWS * HIDN * 4);
    float*           Vb      = (float*)carve((size_t)MROWS * HIDN * 4);
    float*           Ob      = (float*)carve((size_t)MROWS * HIDN * 4);
    __hip_bfloat16*  OH      = (__hip_bfloat16*)carve((size_t)MROWS * HIDN * 2);
    float*           ATT     = (float*)carve((size_t)MROWS * HIDN * 4);
    __hip_bfloat16*  ATTH    = (__hip_bfloat16*)carve((size_t)MROWS * HIDN * 2);
    float*           Rb      = (float*)carve((size_t)MROWS * CDIMN * 4);
    float*           VOTES   = (float*)carve(BN * 9 * 4);
    float*           REPM    = (float*)carve(BN * CDIMN * 4);
    __hip_bfloat16*  FILTA   = (__hip_bfloat16*)carve((size_t)BN * COUTN * CDIMN * 2);
    float*           SMEAN   = (float*)carve(BN * COUTN * 4);
    float*           SVEC    = (float*)carve(BN * COUTN * 4);
    (void)ws_size; (void)in_sizes; (void)n_in; (void)out_size;

    // 1. group-norm stats
    gn_stats_k<<<BN, 256, 0, stream>>>(x, stats);

    // 2. weight casts to bf16
    cast_bf16_k<<<(HIDN * PDIMN + 255) / 256, 256, 0, stream>>>(comp_w, cwh, HIDN * PDIMN);
    cast_bf16_k<<<(HIDN * HIDN + 255) / 256, 256, 0, stream>>>(wq, wqh, HIDN * HIDN);
    cast_bf16_k<<<(HIDN * HIDN + 255) / 256, 256, 0, stream>>>(wk, wkh, HIDN * HIDN);
    cast_bf16_k<<<(HIDN * HIDN + 255) / 256, 256, 0, stream>>>(wv, wvh, HIDN * HIDN);
    cast_bf16_k<<<(HIDN * HIDN + 255) / 256, 256, 0, stream>>>(wo, woh, HIDN * HIDN);
    cast_bf16_k<<<(CDIMN * HIDN + 255) / 256, 256, 0, stream>>>(re_w, rewh, CDIMN * HIDN);

    // 3. input conv -> channels-last bf16 feats
    conv_in_k<<<(BN * HWN) / 256, 256, 0, stream>>>(x, stats, gn_g, gn_b, conv_w, conv_b, featsCL);

    // 4. flatten patches
    flatp_k<<<((size_t)MROWS * PDIMN) / 256, 256, 0, stream>>>(featsCL, flatp);

    // 5. compressor GEMM: E = gelu(flat @ comp_w^T + comp_b)   [3136 x 512, K=24576]
    gemm_nt_k<<<dim3(HIDN / 64, MROWS / 32), 32, 0, stream>>>(flatp, cwh, E, comp_b,
                                                              MROWS, HIDN, PDIMN, 1);
    // 6. LayerNorm + positional tanh
    ln_pos_k<<<MROWS, 256, 0, stream>>>(E, ln_g, ln_b, pos_w, pos_b, HB, HBH);

    // 7. Q/K/V projections (WMMA)
    gemm_nt_k<<<dim3(HIDN / 64, MROWS / 32), 32, 0, stream>>>(HBH, wqh, Qb, bq, MROWS, HIDN, HIDN, 0);
    gemm_nt_k<<<dim3(HIDN / 64, MROWS / 32), 32, 0, stream>>>(HBH, wkh, Kb, bk, MROWS, HIDN, HIDN, 0);
    gemm_nt_k<<<dim3(HIDN / 64, MROWS / 32), 32, 0, stream>>>(HBH, wvh, Vb, bv, MROWS, HIDN, HIDN, 0);

    // 8. attention
    attn_k<<<dim3(SEQN, NHN, BN), 128, 0, stream>>>(Qb, Kb, Vb, Ob);
    cast_bf16_k<<<(MROWS * HIDN + 255) / 256, 256, 0, stream>>>(Ob, OH, MROWS * HIDN);

    // 9. out projection + gelu (WMMA)
    gemm_nt_k<<<dim3(HIDN / 64, MROWS / 32), 32, 0, stream>>>(OH, woh, ATT, bo, MROWS, HIDN, HIDN, 1);
    cast_bf16_k<<<(MROWS * HIDN + 255) / 256, 256, 0, stream>>>(ATT, ATTH, MROWS * HIDN);

    // 10. votes + rep-mean + filter synthesis
    votes_k<<<dim3(9, BN), 256, 0, stream>>>(ATT, be_w, be_b, VOTES);
    gemm_nt_k<<<dim3((CDIMN + 63) / 64, MROWS / 32), 32, 0, stream>>>(ATTH, rewh, Rb, re_b,
                                                                      MROWS, CDIMN, HIDN, 1);
    repm_k<<<(BN * CDIMN + 255) / 256, 256, 0, stream>>>(Rb, REPM);
    filt_k<<<((size_t)BN * COUTN * CDIMN + 255) / 256, 256, 0, stream>>>(REPM, VOTES, patch_basis, FILTA);

    // 11. dynamic conv (implicit WMMA GEMM, 32x64 tiles) + alpha*feats -> y in d_out
    dynconv_k<<<dim3(HWN / 64, COUTN / 32, BN), 32, 0, stream>>>(featsCL, FILTA, alpha, out);

    // 12-14. SE block and final scale
    se_mean_k<<<BN * COUTN, 256, 0, stream>>>(out, SMEAN);
    se_vec_k<<<BN, 128, 0, stream>>>(SMEAN, se_w1, se_b1, se_w2, se_b2, SVEC);
    scale_k<<<((size_t)BN * COUTN * HWN) / 256, 256, 0, stream>>>(out, SVEC);
}
